// FeatureExtractorLayer_91328184582308
// MI455X (gfx1250) — matrix operand order repaired and checked by reading, not compiled, not run on hardware
//
#include <hip/hip_runtime.h>
#include <hip/hip_bf16.h>
#include <stdint.h>

// ---------------------------------------------------------------------------
// Types for CDNA5 WMMA (wave32): v16bf A/B fragments, v8f accumulator.
// ---------------------------------------------------------------------------
typedef __bf16 bf16_t;
typedef __attribute__((ext_vector_type(16))) __bf16 v16bf;
typedef __attribute__((ext_vector_type(8)))  float  v8f;

#define NV_N 100000
#define NC_N 50000
#define NE_N 500000

// ---------------------------------------------------------------------------
// Tiled GEMM: out[M,N] = relu?( A[M,Kp](bf16) @ W[Kp,N](bf16) + bias )
// One wave per 16xN output strip (NT = N/16 accumulators sharing one A frag).
// K-loop in steps of 32 via v_wmma_f32_16x16x32_bf16.
// A is row-major bf16 (Kp multiple of 32, zero padded).
// W is PRE-SWIZZLED into per-lane-contiguous fragment order:
//   Wsw[(((kt*NT + nt)*32 + lane)*16 + j]  with
//   lane<16 : col = nt*16+lane,     K = kt*32 + j
//   lane>=16: col = nt*16+lane-16,  K = kt*32 + 16 + j
// so each lane's B fragment is two b128 loads.  (contiguous K order, same as the A fragment.)
// ---------------------------------------------------------------------------
template<int NT, int RELU>
__global__ __launch_bounds__(256) void gemm_bf16_wmma(
    const bf16_t* __restrict__ A, const bf16_t* __restrict__ Wsw,
    const float* __restrict__ bias, float* __restrict__ out,
    int M, int Kp)
{
    constexpr int N = NT * 16;
    int wave = (int)(threadIdx.x >> 5) + blockIdx.x * 8;
    if (wave >= (M >> 4)) return;            // uniform per wave: EXEC stays full

    int lane = threadIdx.x & 31;
    int half = lane >> 4;                    // 0: lanes 0-15, 1: lanes 16-31
    int l16  = lane & 15;

    const bf16_t* arow = A + (size_t)(wave * 16 + l16) * (size_t)Kp;

    v8f acc[NT];
#pragma unroll
    for (int nt = 0; nt < NT; ++nt) acc[nt] = (v8f){};

    int ktiles = Kp >> 5;
    for (int kt = 0; kt < ktiles; ++kt) {
        v16bf a;
        int ka = kt * 32 + half * 16;        // A: K = ka..ka+15
#pragma unroll
        for (int i = 0; i < 8; ++i) a[i]     = arow[ka + i];
#pragma unroll
        for (int i = 0; i < 8; ++i) a[i + 8] = arow[ka + 8 + i];

        const bf16_t* wp = Wsw + ((size_t)(kt * NT) * 32 + lane) * 16;
#pragma unroll
        for (int nt = 0; nt < NT; ++nt) {
            v16bf b;
            const bf16_t* bp = wp + (size_t)nt * (32 * 16);
#pragma unroll
            for (int i = 0; i < 16; ++i) b[i] = bp[i];   // 2x b128, contiguous
            acc[nt] = __builtin_amdgcn_wmma_f32_16x16x32_bf16(
                          false, a, false, b, (short)0, acc[nt], false, false);
        }
    }

#pragma unroll
    for (int nt = 0; nt < NT; ++nt) {
        int ncol = nt * 16 + l16;
        float bv = bias ? bias[ncol] : 0.0f;
        size_t baseo = (size_t)(wave * 16 + half * 8) * (size_t)N + ncol;
#pragma unroll
        for (int r = 0; r < 8; ++r) {        // D: VGPR r -> row r (+8 hi half)
            float v = acc[nt][r] + bv;
            if (RELU) v = fmaxf(v, 0.0f);
            out[baseo + (size_t)r * (size_t)N] = v;
        }
    }
}

// ---------------------------------------------------------------------------
// Packing kernels (f32 -> zero-padded bf16 matrices for WMMA)
// ---------------------------------------------------------------------------
__global__ void pack_nodes(const float* __restrict__ learned,
                           const float* __restrict__ lp,
                           bf16_t* __restrict__ dst, int n_nodes)
{
    int i = blockIdx.x * blockDim.x + threadIdx.x;
    int total = n_nodes * 96;
    if (i >= total) return;
    int n = i / 96, c = i - n * 96;
    float v = (c < 64) ? learned[(size_t)n * 64 + c]
            : (c < 72) ? lp[(size_t)n * 8 + (c - 64)]
            : 0.0f;
    dst[i] = (bf16_t)v;
}

__global__ void pack_edges(const float* __restrict__ elearn,
                           const float* __restrict__ lo,
                           const float* __restrict__ hi,
                           const float* __restrict__ dmm,
                           const float* __restrict__ rest,
                           bf16_t* __restrict__ dst, int n_edges)
{
    int i = blockIdx.x * blockDim.x + threadIdx.x;
    int total = n_edges * 64;
    if (i >= total) return;
    int e = i >> 6, c = i & 63;
    float v;
    if (c < 32)      v = elearn[(size_t)e * 32 + c];
    else if (c == 32) v = lo[e];
    else if (c == 33) v = hi[e];
    else if (c == 34) v = dmm[e];
    else if (c < 40)  v = rest[(size_t)e * 5 + (c - 35)];
    else              v = 0.0f;
    dst[i] = (bf16_t)v;
}

// Pack weight [K,N] f32 -> swizzled bf16 fragment order (see gemm comment).
__global__ void pack_weight_sw(const float* __restrict__ W, bf16_t* __restrict__ Wsw,
                               int K, int N, int Kp)
{
    int i = blockIdx.x * blockDim.x + threadIdx.x;
    if (i >= Kp * N) return;
    int j    = i & 15;
    int lane = (i >> 4) & 31;
    int t    = i >> 9;                     // kt*NT + nt
    int NTl  = N >> 4;
    int kt   = t / NTl;
    int nt   = t - kt * NTl;
    int n = nt * 16 + (lane & 15);
    int k = kt * 32 + (lane >> 4) * 16 + j;
    Wsw[i] = (k < K) ? (bf16_t)W[(size_t)k * N + n] : (bf16_t)0.0f;
}

__global__ void f32_to_bf16_k(const float* __restrict__ a, bf16_t* __restrict__ b, int n)
{
    int i = blockIdx.x * blockDim.x + threadIdx.x;
    if (i < n) b[i] = (bf16_t)a[i];
}

// ---------------------------------------------------------------------------
// Attention scatter: one wave32 per edge.  logit = scale * q[dst].(k[src]+ee)
// w = exp(logit);  atomic-accumulate w, count, and w*(v[src]+ee) per dst.
// q/k/v tables live in L2 (192MB); atomics resolve in L2.
// ---------------------------------------------------------------------------
__global__ __launch_bounds__(256) void att_scatter(
    const float* __restrict__ q, const float* __restrict__ k,
    const float* __restrict__ v, const float* __restrict__ ee,
    const int* __restrict__ src, const int* __restrict__ dst,
    float* __restrict__ ssum, float* __restrict__ deg,
    float* __restrict__ agg, int n_edges)
{
    int wave = (int)(threadIdx.x >> 5) + blockIdx.x * 8;
    if (wave >= n_edges) return;
    int lane = threadIdx.x & 31;
    int s = src[wave], d = dst[wave];

    const float* qr = q + (size_t)d * 64;
    const float* kr = k + (size_t)s * 64;
    const float* vr = v + (size_t)s * 64;
    const float* er = ee + (size_t)wave * 64;

    float e0 = er[lane], e1 = er[lane + 32];
    float p = qr[lane] * (kr[lane] + e0) + qr[lane + 32] * (kr[lane + 32] + e1);
#pragma unroll
    for (int off = 16; off > 0; off >>= 1) p += __shfl_xor(p, off, 32);

    float w = __expf(fminf(p * 0.125f, 30.0f));   // scale = 1/sqrt(64)
    if (lane == 0) {
        atomicAdd(ssum + d, w);
        atomicAdd(deg + d, 1.0f);
    }
    atomicAdd(agg + (size_t)d * 64 + lane,      (vr[lane] + e0) * w);
    atomicAdd(agg + (size_t)d * 64 + lane + 32, (vr[lane + 32] + e1) * w);
}

// ---------------------------------------------------------------------------
// Node finalize: x = relu(agg/(sum*max(deg,1)) + skip) -> f32 out; also build
// the bf16 [*,96] combined feature matrix ([x | lp | 0-pad]) for next GEMMs.
// ---------------------------------------------------------------------------
__global__ void finalize_node(const float* __restrict__ agg,
                              const float* __restrict__ ssum,
                              const float* __restrict__ deg,
                              const float* __restrict__ skip,
                              const float* __restrict__ lp,
                              float* __restrict__ out,
                              bf16_t* __restrict__ comb2, int n_nodes)
{
    int i = blockIdx.x * blockDim.x + threadIdx.x;
    if (i >= n_nodes * 64) return;
    int n = i >> 6, ch = i & 63;
    float sden = fmaxf(ssum[n], 1e-16f);
    float dden = fmaxf(deg[n], 1.0f);
    float val = fmaxf(agg[i] / (sden * dden) + skip[i], 0.0f);
    out[i] = val;
    bf16_t* row = comb2 + (size_t)n * 96;
    row[ch] = (bf16_t)val;
    if (ch < 8)  row[64 + ch] = (bf16_t)lp[(size_t)n * 8 + ch];
    if (ch < 24) row[72 + ch] = (bf16_t)0.0f;
}

// ---------------------------------------------------------------------------
// Build edge-MLP input: [edge_comb(40) | vc[evar](32) | cc[econ](32) | pad] bf16
// ---------------------------------------------------------------------------
__global__ void build_ein(const bf16_t* __restrict__ Aedge,
                          const float* __restrict__ vc,
                          const float* __restrict__ cc,
                          const int* __restrict__ evar,
                          const int* __restrict__ econ,
                          bf16_t* __restrict__ ein, int n_edges)
{
    long long i = (long long)blockIdx.x * blockDim.x + threadIdx.x;
    long long total = (long long)n_edges * 128;
    if (i >= total) return;
    long long e = i >> 7;
    int c = (int)(i & 127);
    bf16_t v;
    if (c < 40)       v = Aedge[e * 64 + c];
    else if (c < 72)  v = (bf16_t)vc[(size_t)evar[e] * 32 + (c - 40)];
    else if (c < 104) v = (bf16_t)cc[(size_t)econ[e] * 32 + (c - 72)];
    else              v = (bf16_t)0.0f;
    ein[i] = v;
}

// ---------------------------------------------------------------------------
// Host side
// ---------------------------------------------------------------------------
static inline int cdiv(long long a, long long b) { return (int)((a + b - 1) / b); }

static inline void launch_gemm(const bf16_t* A, const bf16_t* Wsw, const float* bias,
                               float* out, int M, int Kp, int N, int relu,
                               hipStream_t s)
{
    int blocks = cdiv(M / 16, 8);
    if (N == 64) {
        if (relu) gemm_bf16_wmma<4, 1><<<blocks, 256, 0, s>>>(A, Wsw, bias, out, M, Kp);
        else      gemm_bf16_wmma<4, 0><<<blocks, 256, 0, s>>>(A, Wsw, bias, out, M, Kp);
    } else {
        if (relu) gemm_bf16_wmma<2, 1><<<blocks, 256, 0, s>>>(A, Wsw, bias, out, M, Kp);
        else      gemm_bf16_wmma<2, 0><<<blocks, 256, 0, s>>>(A, Wsw, bias, out, M, Kp);
    }
}

extern "C" void kernel_launch(void* const* d_in, const int* in_sizes, int n_in,
                              void* d_out, int out_size, void* d_ws, size_t ws_size,
                              hipStream_t stream)
{
    (void)in_sizes; (void)n_in; (void)out_size; (void)ws_size;

    const float* var_learned = (const float*)d_in[0];
    const float* var_lp      = (const float*)d_in[1];
    const float* con_learned = (const float*)d_in[2];
    const float* con_lp      = (const float*)d_in[3];
    const float* edge_learn  = (const float*)d_in[4];
    const float* lo          = (const float*)d_in[5];
    const float* hi          = (const float*)d_in[6];
    const float* dmm         = (const float*)d_in[7];
    const float* erest       = (const float*)d_in[8];
    const int*   evar        = (const int*)d_in[9];
    const int*   econ        = (const int*)d_in[10];
    const float* cuWq = (const float*)d_in[11]; const float* cubq = (const float*)d_in[12];
    const float* cuWk = (const float*)d_in[13]; const float* cubk = (const float*)d_in[14];
    const float* cuWv = (const float*)d_in[15]; const float* cubv = (const float*)d_in[16];
    const float* cuWe = (const float*)d_in[17];
    const float* cuWs = (const float*)d_in[18]; const float* cubs = (const float*)d_in[19];
    const float* vuWq = (const float*)d_in[20]; const float* vubq = (const float*)d_in[21];
    const float* vuWk = (const float*)d_in[22]; const float* vubk = (const float*)d_in[23];
    const float* vuWv = (const float*)d_in[24]; const float* vubv = (const float*)d_in[25];
    const float* vuWe = (const float*)d_in[26];
    const float* vuWs = (const float*)d_in[27]; const float* vubs = (const float*)d_in[28];
    const float* evW1 = (const float*)d_in[29]; const float* evb1 = (const float*)d_in[30];
    const float* evW2 = (const float*)d_in[31]; const float* evb2 = (const float*)d_in[32];
    const float* ecW1 = (const float*)d_in[33]; const float* ecb1 = (const float*)d_in[34];
    const float* ecW2 = (const float*)d_in[35]; const float* ecb2 = (const float*)d_in[36];
    const float* eeW1 = (const float*)d_in[37]; const float* eeb1 = (const float*)d_in[38];
    const float* eeW2 = (const float*)d_in[39]; const float* eeb2 = (const float*)d_in[40];

    float* out      = (float*)d_out;
    float* var_out  = out;
    float* con_out  = out + (size_t)NV_N * 64;
    float* edge_out = out + (size_t)NV_N * 64 + (size_t)NC_N * 64;

    // ---- workspace bump allocator -----------------------------------------
    uint8_t* base = (uint8_t*)d_ws;
    size_t off = 0;
    auto alloc = [&](size_t bytes) -> void* {
        void* p = base + off;
        off = (off + bytes + 255) & ~(size_t)255;
        return p;
    };
    bf16_t* Avar   = (bf16_t*)alloc((size_t)NV_N * 96 * 2);
    bf16_t* Acon   = (bf16_t*)alloc((size_t)NC_N * 96 * 2);
    bf16_t* Aedge  = (bf16_t*)alloc((size_t)NE_N * 64 * 2);
    bf16_t* Ccomb2 = (bf16_t*)alloc((size_t)NC_N * 96 * 2);
    bf16_t* Vcomb2 = (bf16_t*)alloc((size_t)NV_N * 96 * 2);
    // packed (swizzled) weights
    bf16_t* pcuWq = (bf16_t*)alloc(96 * 64 * 2);
    bf16_t* pcuWk = (bf16_t*)alloc(96 * 64 * 2);
    bf16_t* pcuWv = (bf16_t*)alloc(96 * 64 * 2);
    bf16_t* pcuWs = (bf16_t*)alloc(96 * 64 * 2);
    bf16_t* pcuWe = (bf16_t*)alloc(64 * 64 * 2);
    bf16_t* pvuWq = (bf16_t*)alloc(96 * 64 * 2);
    bf16_t* pvuWk = (bf16_t*)alloc(96 * 64 * 2);
    bf16_t* pvuWv = (bf16_t*)alloc(96 * 64 * 2);
    bf16_t* pvuWs = (bf16_t*)alloc(96 * 64 * 2);
    bf16_t* pvuWe = (bf16_t*)alloc(64 * 64 * 2);
    bf16_t* pevW1 = (bf16_t*)alloc(96 * 32 * 2);
    bf16_t* pecW1 = (bf16_t*)alloc(96 * 32 * 2);
    bf16_t* pevW2 = (bf16_t*)alloc(32 * 32 * 2);
    bf16_t* pecW2 = (bf16_t*)alloc(32 * 32 * 2);
    bf16_t* peeW1 = (bf16_t*)alloc(128 * 32 * 2);
    bf16_t* peeW2 = (bf16_t*)alloc(32 * 32 * 2);
    // f32 intermediates
    float* q_c    = (float*)alloc((size_t)NC_N * 64 * 4);
    float* skip_c = (float*)alloc((size_t)NC_N * 64 * 4);
    float* k_v    = (float*)alloc((size_t)NV_N * 64 * 4);
    float* v_v    = (float*)alloc((size_t)NV_N * 64 * 4);
    float* q_v    = (float*)alloc((size_t)NV_N * 64 * 4);
    float* skip_v = (float*)alloc((size_t)NV_N * 64 * 4);
    float* k_c    = (float*)alloc((size_t)NC_N * 64 * 4);
    float* v_c    = (float*)alloc((size_t)NC_N * 64 * 4);
    float* ee     = (float*)alloc((size_t)NE_N * 64 * 4);   // reused cu then vu
    float* s_c    = (float*)alloc((size_t)NC_N * 4);
    float* deg_c  = (float*)alloc((size_t)NC_N * 4);
    float* agg_c  = (float*)alloc((size_t)NC_N * 64 * 4);
    float* s_v    = (float*)alloc((size_t)NV_N * 4);
    float* deg_v  = (float*)alloc((size_t)NV_N * 4);
    float* agg_v  = (float*)alloc((size_t)NV_N * 64 * 4);
    float* vh     = (float*)alloc((size_t)NV_N * 32 * 4);
    bf16_t* vh_bf = (bf16_t*)alloc((size_t)NV_N * 32 * 2);
    float* vc     = (float*)alloc((size_t)NV_N * 32 * 4);
    float* ch     = (float*)alloc((size_t)NC_N * 32 * 4);
    bf16_t* ch_bf = (bf16_t*)alloc((size_t)NC_N * 32 * 2);
    float* cc     = (float*)alloc((size_t)NC_N * 32 * 4);
    bf16_t* Ein   = (bf16_t*)alloc((size_t)NE_N * 128 * 2);
    float* eh     = (float*)alloc((size_t)NE_N * 32 * 4);
    bf16_t* eh_bf = (bf16_t*)alloc((size_t)NE_N * 32 * 2);

    // ---- stage 0: pack inputs / weights to bf16 ---------------------------
    pack_nodes<<<cdiv((long long)NV_N * 96, 256), 256, 0, stream>>>(var_learned, var_lp, Avar, NV_N);
    pack_nodes<<<cdiv((long long)NC_N * 96, 256), 256, 0, stream>>>(con_learned, con_lp, Acon, NC_N);
    pack_edges<<<cdiv((long long)NE_N * 64, 256), 256, 0, stream>>>(edge_learn, lo, hi, dmm, erest, Aedge, NE_N);

    struct { const float* w; bf16_t* p; int K, N, Kp; } wts[] = {
        {cuWq, pcuWq, 72, 64, 96}, {cuWk, pcuWk, 72, 64, 96}, {cuWv, pcuWv, 72, 64, 96},
        {cuWs, pcuWs, 72, 64, 96}, {cuWe, pcuWe, 40, 64, 64},
        {vuWq, pvuWq, 72, 64, 96}, {vuWk, pvuWk, 72, 64, 96}, {vuWv, pvuWv, 72, 64, 96},
        {vuWs, pvuWs, 72, 64, 96}, {vuWe, pvuWe, 40, 64, 64},
        {evW1, pevW1, 72, 32, 96}, {ecW1, pecW1, 72, 32, 96},
        {evW2, pevW2, 32, 32, 32}, {ecW2, pecW2, 32, 32, 32},
        {eeW1, peeW1, 104, 32, 128}, {eeW2, peeW2, 32, 32, 32},
    };
    for (auto& t : wts)
        pack_weight_sw<<<cdiv(t.Kp * t.N, 256), 256, 0, stream>>>(t.w, t.p, t.K, t.N, t.Kp);

    // ---- stage 1: con updater TransformerConv -----------------------------
    launch_gemm(Acon,  pcuWq, cubq, q_c,    NC_N, 96, 64, 0, stream);
    launch_gemm(Acon,  pcuWs, cubs, skip_c, NC_N, 96, 64, 0, stream);
    launch_gemm(Avar,  pcuWk, cubk, k_v,    NV_N, 96, 64, 0, stream);
    launch_gemm(Avar,  pcuWv, cubv, v_v,    NV_N, 96, 64, 0, stream);
    launch_gemm(Aedge, pcuWe, nullptr, ee,  NE_N, 64, 64, 0, stream);

    hipMemsetAsync(s_c,   0, (size_t)NC_N * 4,      stream);
    hipMemsetAsync(deg_c, 0, (size_t)NC_N * 4,      stream);
    hipMemsetAsync(agg_c, 0, (size_t)NC_N * 64 * 4, stream);
    att_scatter<<<cdiv(NE_N, 8), 256, 0, stream>>>(q_c, k_v, v_v, ee, evar, econ,
                                                   s_c, deg_c, agg_c, NE_N);
    finalize_node<<<cdiv((long long)NC_N * 64, 256), 256, 0, stream>>>(
        agg_c, s_c, deg_c, skip_c, con_lp, con_out, Ccomb2, NC_N);

    // ---- stage 2: var updater TransformerConv -----------------------------
    launch_gemm(Avar,   pvuWq, vubq, q_v,    NV_N, 96, 64, 0, stream);
    launch_gemm(Avar,   pvuWs, vubs, skip_v, NV_N, 96, 64, 0, stream);
    launch_gemm(Ccomb2, pvuWk, vubk, k_c,    NC_N, 96, 64, 0, stream);
    launch_gemm(Ccomb2, pvuWv, vubv, v_c,    NC_N, 96, 64, 0, stream);
    launch_gemm(Aedge,  pvuWe, nullptr, ee,  NE_N, 64, 64, 0, stream);

    hipMemsetAsync(s_v,   0, (size_t)NV_N * 4,      stream);
    hipMemsetAsync(deg_v, 0, (size_t)NV_N * 4,      stream);
    hipMemsetAsync(agg_v, 0, (size_t)NV_N * 64 * 4, stream);
    att_scatter<<<cdiv(NE_N, 8), 256, 0, stream>>>(q_v, k_c, v_c, ee, econ, evar,
                                                   s_v, deg_v, agg_v, NE_N);
    finalize_node<<<cdiv((long long)NV_N * 64, 256), 256, 0, stream>>>(
        agg_v, s_v, deg_v, skip_v, var_lp, var_out, Vcomb2, NV_N);

    // ---- stage 3: EdgeUpdater MLPs ----------------------------------------
    launch_gemm(Vcomb2, pevW1, evb1, vh, NV_N, 96, 32, 1, stream);
    f32_to_bf16_k<<<cdiv((long long)NV_N * 32, 256), 256, 0, stream>>>(vh, vh_bf, NV_N * 32);
    launch_gemm(vh_bf,  pevW2, evb2, vc, NV_N, 32, 32, 1, stream);

    launch_gemm(Ccomb2, pecW1, ecb1, ch, NC_N, 96, 32, 1, stream);
    f32_to_bf16_k<<<cdiv((long long)NC_N * 32, 256), 256, 0, stream>>>(ch, ch_bf, NC_N * 32);
    launch_gemm(ch_bf,  pecW2, ecb2, cc, NC_N, 32, 32, 1, stream);

    build_ein<<<cdiv((long long)NE_N * 128, 256), 256, 0, stream>>>(
        Aedge, vc, cc, evar, econ, Ein, NE_N);

    launch_gemm(Ein,   peeW1, eeb1, eh, NE_N, 128, 32, 1, stream);
    f32_to_bf16_k<<<cdiv((long long)NE_N * 32, 256), 256, 0, stream>>>(eh, eh_bf, NE_N * 32);
    launch_gemm(eh_bf, peeW2, eeb2, edge_out, NE_N, 32, 32, 1, stream);
}
